// TreeANFIS_25426206392905
// MI455X (gfx1250) — matrix operand order repaired and checked by reading, not compiled, not run on hardware
//
#include <hip/hip_runtime.h>
#include <hip/hip_bf16.h>

typedef __attribute__((ext_vector_type(16))) __bf16 v16bf;
typedef __attribute__((ext_vector_type(8)))  float  v8f;

#define F_DIM 64
#define R_DIM 1024
#define KD    256      // GEMM K-dim = consequent cols 1..256
#define CROW  257
#define RHALF 512      // rules per pass (firing buffer reuse)

__device__ __forceinline__ unsigned short f32_to_bf16(float f) {
  union { float f; unsigned u; } c; c.f = f;
  unsigned u = c.u;
  u += 0x7FFFu + ((u >> 16) & 1u);          // round-to-nearest-even
  return (unsigned short)(u >> 16);
}

// Pre-swizzle consequent_params[:,1:257] into bf16 WMMA B-fragment layout.
// B (32x16 per k-step): lane = n + 16*((k/8)%2), element e = k%8 + 8*(k/16).
// cb flat index = ((rtile*8 + kb)*32 + lane)*16 + e
__global__ void prep_cb_kernel(const float* __restrict__ cons,
                               unsigned short* __restrict__ cb) {
  int i = blockIdx.x * 256 + threadIdx.x;   // 0 .. 262143
  int e     =  i        & 15;
  int lane  = (i >> 4)  & 31;
  int kb    = (i >> 9)  & 7;
  int rtile =  i >> 12;                     // 0..63
  int r  = rtile * 16 + (lane & 15);        // rule index (N)
  int kl = (e & 7) + 8 * (lane >> 4) + 16 * (e >> 3);   // K within the 32-block
  cb[i] = f32_to_bf16(cons[r * CROW + 1 + kb * 32 + kl]);
}

__device__ __forceinline__ float fuzzy_leg(float xv, float t, float ps, float m) {
  float z  = ps * (xv - t);                      // premise*sign*(sel-thresh)
  float e  = __expf(-z);                         // v_exp_f32
  float mf = __builtin_amdgcn_rcpf(1.0f + e);    // sigmoid via v_rcp_f32
  return fmaf(m, mf - 1.0f, 1.0f);               // mask? mf : 1
}

// Fused epilogue: part[v] += firing[b=v+8h, r] * (D[v] + bias[r])
#define EPILOGUE(ACC, RTL, PASS)                                          \
  {                                                                       \
    int rr_ = (RTL) * 16 + mrow;                                          \
    float biasr_ = bias_lds[(PASS) * RHALF + rr_];                        \
    const float4* fq_ = (const float4*)&f_lds[rr_ * 16 + 8 * h];          \
    float4 fa_ = fq_[0], fb_ = fq_[1];                                    \
    part[0] = fmaf(fa_.x, (ACC)[0] + biasr_, part[0]);                    \
    part[1] = fmaf(fa_.y, (ACC)[1] + biasr_, part[1]);                    \
    part[2] = fmaf(fa_.z, (ACC)[2] + biasr_, part[2]);                    \
    part[3] = fmaf(fa_.w, (ACC)[3] + biasr_, part[3]);                    \
    part[4] = fmaf(fb_.x, (ACC)[4] + biasr_, part[4]);                    \
    part[5] = fmaf(fb_.y, (ACC)[5] + biasr_, part[5]);                    \
    part[6] = fmaf(fb_.z, (ACC)[6] + biasr_, part[6]);                    \
    part[7] = fmaf(fb_.w, (ACC)[7] + biasr_, part[7]);                    \
  }

__global__ void __launch_bounds__(128)
anfis_kernel(const float* __restrict__ x,
             const float* __restrict__ aw,
             const float* __restrict__ prem,
             const float* __restrict__ th,
             const float* __restrict__ sg,
             const float* __restrict__ mk,
             const float* __restrict__ cons,
             const int*   __restrict__ fidx,
             const int*   __restrict__ pairs,
             const unsigned short* __restrict__ cb,
             float* __restrict__ out) {
  __shared__ float xa_lds[16 * F_DIM];                       // 4 KB
  __shared__ __align__(16) unsigned short p_lds[16 * KD];    // 8 KB bf16 p-tile
  __shared__ __align__(16) float f_lds[RHALF * 16];          // 32 KB firing [lr][b], reused per pass
  __shared__ float bias_lds[R_DIM];                          // 4 KB
  __shared__ float den_lds[16];
  __shared__ float num_lds[16];

  const int tid = threadIdx.x;
  const int b0  = blockIdx.x * 16;

  // ---- phase 1: attention-weighted x tile; zero reducers ----
  #pragma unroll
  for (int j = 0; j < 8; ++j) {
    int i = tid + 128 * j;                  // i = b*64 + f
    int bb = i >> 6, f = i & 63;
    xa_lds[i] = x[(b0 + bb) * F_DIM + f] * aw[f];
  }
  if (tid < 16) { den_lds[tid] = 0.0f; num_lds[tid] = 0.0f; }
  __syncthreads();

  // ---- phase 2a: p feature tile in bf16 (A-matrix source) ----
  #pragma unroll
  for (int j = 0; j < 32; ++j) {
    int i = tid + 128 * j;                  // i = m*256 + k
    int m = i >> 8, k = i & 255;
    float v;
    if (k < 64) {
      v = xa_lds[m * 64 + k];
    } else if (k < 128) {
      float t = xa_lds[m * 64 + (k - 64)];
      v = t * t;
    } else {
      int pp = k - 128;
      v = xa_lds[m * 64 + pairs[pp * 2]] * xa_lds[m * 64 + pairs[pp * 2 + 1]];
    }
    p_lds[i] = f32_to_bf16(v);
  }
  // ---- phase 2b: bias column (consequent col 0) ----
  #pragma unroll
  for (int j = 0; j < 8; ++j) {
    int r = tid + 128 * j;
    bias_lds[r] = cons[r * CROW];
  }

  const int lane = tid & 31;
  const int wave = tid >> 5;
  const int mrow = lane & 15;       // A-matrix M row / D-tile N helper
  const int h    = lane >> 4;
  const uint4* cbq = (const uint4*)cb;

  // ==== two passes of 512 rules, reusing the 32 KB firing buffer ====
  for (int pass = 0; pass < 2; ++pass) {
    // ---- firing strengths: 4 rules per thread x 16 batch rows ----
    {
      float denacc[16];
      #pragma unroll
      for (int bb = 0; bb < 16; ++bb) denacc[bb] = 0.0f;
      #pragma unroll 1
      for (int j = 0; j < 4; ++j) {
        int lr = tid * 4 + j;               // local rule 0..511
        int r  = pass * RHALF + lr;
        int i0 = fidx[r * 4 + 0], i1 = fidx[r * 4 + 1];
        int i2 = fidx[r * 4 + 2], i3 = fidx[r * 4 + 3];
        float pr = prem[r];
        float p0 = pr * sg[r * 4 + 0], p1 = pr * sg[r * 4 + 1];
        float p2 = pr * sg[r * 4 + 2], p3 = pr * sg[r * 4 + 3];
        float t0 = th[r * 4 + 0], t1 = th[r * 4 + 1];
        float t2 = th[r * 4 + 2], t3 = th[r * 4 + 3];
        float m0 = mk[r * 4 + 0], m1 = mk[r * 4 + 1];
        float m2 = mk[r * 4 + 2], m3 = mk[r * 4 + 3];
        #pragma unroll
        for (int bb = 0; bb < 16; ++bb) {
          const float* xa = &xa_lds[bb * 64];
          float f = fuzzy_leg(xa[i0], t0, p0, m0)
                  * fuzzy_leg(xa[i1], t1, p1, m1)
                  * fuzzy_leg(xa[i2], t2, p2, m2)
                  * fuzzy_leg(xa[i3], t3, p3, m3);
          f_lds[lr * 16 + bb] = f;
          denacc[bb] += f;
        }
      }
      #pragma unroll
      for (int bb = 0; bb < 16; ++bb) atomicAdd(&den_lds[bb], denacc[bb]);
    }
    __syncthreads();   // firing (and, on pass 0, p-tile/bias) ready

    // ---- WMMA readout: this wave handles 8 rule tiles, 2 at a time ----
    // A fragments (16-bit A 16x32): lane L: M=L%16; e<8 -> K=e+8h; e>=8 -> K=e+8+8h
    v16bf afr[8];
    #pragma unroll
    for (int kb = 0; kb < 8; ++kb) {
      const uint4* pq = (const uint4*)&p_lds[mrow * KD + kb * 32 + 8 * h];
      union { v16bf v; uint4 q[2]; } A;
      A.q[0] = pq[0];     // K = base .. base+7
      A.q[1] = pq[2];     // K = base+16 .. base+23
      afr[kb] = A.v;
    }

    float part[8];
    #pragma unroll
    for (int v = 0; v < 8; ++v) part[v] = 0.0f;

    #pragma unroll 1
    for (int t = 0; t < 8; t += 2) {        // 2 rule tiles / iter: dual WMMA chains
      int rtl0 = wave * 8 + t;              // local tile in this pass
      int rtl1 = rtl0 + 1;
      int rt0  = pass * 32 + rtl0;          // global tile (cb layout)
      int rt1  = rt0 + 1;
      v8f acc0 = {0.f, 0.f, 0.f, 0.f, 0.f, 0.f, 0.f, 0.f};
      v8f acc1 = {0.f, 0.f, 0.f, 0.f, 0.f, 0.f, 0.f, 0.f};
      #pragma unroll
      for (int kb = 0; kb < 8; ++kb) {
        union { v16bf v; uint4 q[2]; } B0, B1;
        int base0 = ((rt0 * 8 + kb) * 32 + lane) * 2;   // uint4 units
        int base1 = ((rt1 * 8 + kb) * 32 + lane) * 2;
        B0.q[0] = cbq[base0];  B0.q[1] = cbq[base0 + 1];
        B1.q[0] = cbq[base1];  B1.q[1] = cbq[base1 + 1];
        acc0 = __builtin_amdgcn_wmma_f32_16x16x32_bf16(
            false, afr[kb], false, B0.v, (short)0, acc0, false, false);
        acc1 = __builtin_amdgcn_wmma_f32_16x16x32_bf16(
            false, afr[kb], false, B1.v, (short)0, acc1, false, false);
      }
      EPILOGUE(acc0, rtl0, pass)
      EPILOGUE(acc1, rtl1, pass)
    }
    #pragma unroll
    for (int v = 0; v < 8; ++v) atomicAdd(&num_lds[v + 8 * h], part[v]);
    __syncthreads();   // pass results committed before f_lds reuse / final read
  }

  // ---- deferred normalization and store ----
  if (tid < 16) out[b0 + tid] = num_lds[tid] / (den_lds[tid] + 1e-8f);
}

extern "C" void kernel_launch(void* const* d_in, const int* in_sizes, int n_in,
                              void* d_out, int out_size, void* d_ws, size_t ws_size,
                              hipStream_t stream) {
  const float* x     = (const float*)d_in[0];
  const float* aw    = (const float*)d_in[1];
  const float* prem  = (const float*)d_in[2];
  const float* th    = (const float*)d_in[3];
  const float* sg    = (const float*)d_in[4];
  const float* mk    = (const float*)d_in[5];
  const float* cons  = (const float*)d_in[6];
  const int*   fidx  = (const int*)d_in[7];
  const int*   pairs = (const int*)d_in[8];

  unsigned short* cb = (unsigned short*)d_ws;   // 512 KB bf16 swizzled C

  // Pre-swizzle C into WMMA B-fragment layout (recomputed every call: deterministic).
  prep_cb_kernel<<<1024, 256, 0, stream>>>(cons, cb);

  // One workgroup per 16 batch rows: 16384/16 = 1024 blocks of 4 wave32s.
  anfis_kernel<<<1024, 128, 0, stream>>>(x, aw, prem, th, sg, mk, cons,
                                         fidx, pairs, cb, (float*)d_out);
}
